// SBDolomiteBlock_10754598109377
// MI455X (gfx1250) — compile-verified, matrix operand order checked by reading, not run on hardware
//
#include <hip/hip_runtime.h>
#include <hip/hip_bf16.h>
#include <math.h>

// ---------------------------------------------------------------------------
// SB-Dolomite transformer block on gfx1250 (MI455X), f16 WMMA everywhere.
// T=2048 tokens, HID=1024, H=16 heads, HD=64, INNER=4096.
// GEMMs: 128x128 block tile, 32x64 wave tile, K-step 32, double-buffered LDS,
// A tiles staged by the Tensor Data Mover (TENSOR_LOAD_TO_LDS + TENSORcnt).
// ---------------------------------------------------------------------------

#define T_TOK  2048
#define HID    1024
#define NHEAD  16
#define HD     64
#define INNER  4096
#define QKVW   (3 * HID)   // 3072

typedef __attribute__((ext_vector_type(16))) _Float16 v16h;
typedef __attribute__((ext_vector_type(8)))  float    v8f;
typedef __attribute__((ext_vector_type(4)))  unsigned int u32x4;
typedef __attribute__((ext_vector_type(8)))  int          i32x8;
typedef __attribute__((ext_vector_type(4)))  int          i32x4;

#ifndef __has_builtin
#define __has_builtin(x) 0
#endif

#if __has_builtin(__builtin_amdgcn_tensor_load_to_lds) && \
    __has_builtin(__builtin_amdgcn_s_wait_tensorcnt)
#define USE_TDM 1
#else
#define USE_TDM 0
#endif

// ---------------------------------------------------------------------------
// WMMA helpers (wave32, 16x16x32 f16 -> f32)
// ---------------------------------------------------------------------------
__device__ __forceinline__ v8f wmma_f16(v16h a, v16h b, v8f c) {
  return __builtin_amdgcn_wmma_f32_16x16x32_f16(
      /*neg_a=*/false, a, /*neg_b=*/false, b,
      /*c_mod=*/(short)0, c, /*reuse_a=*/false, /*reuse_b=*/false);
}

// K index for half-pair j (0..7) given lane-half hi (0/1), per ISA 16-bit
// 16x32 A-matrix layout (B uses the same layout on a transposed tile).
__device__ __forceinline__ int kpair(int j, int hi) {
  return (j < 4) ? (hi * 8 + 2 * j) : (16 + hi * 8 + 2 * (j - 4));
}

// Load a 16x32 fragment from an LDS tile with row stride 32 halves.
__device__ __forceinline__ v16h frag_ld_lds(const _Float16* base, int row, int lane) {
  const int m = lane & 15, hi = lane >> 4;
  const _Float16* p = base + (size_t)(row + m) * 32;
  v16h a;
#pragma unroll
  for (int j = 0; j < 8; ++j) {
    const int k = kpair(j, hi);
    a[2 * j]     = p[k];
    a[2 * j + 1] = p[k + 1];
  }
  return a;
}

__device__ __forceinline__ float gelu_tanh(float x) {
  const float x3 = x * x * x;
  return 0.5f * x * (1.0f + tanhf(0.7978845608028654f * (x + 0.044715f * x3)));
}

#if USE_TDM
// ---------------------------------------------------------------------------
// Tensor Data Mover: async 2D tile load (16-bit elements) global -> LDS.
// D# packing per CDNA5 ISA 8.3/8.4 (group0: count/lds/global/type;
// group1: data_size, tensor_dim0/1, tile_dim0/1, dim0_stride). Groups 2/3
// zero (2D tile: tile_dim2=0 -> unused). Row-major fill matches LDS tile.
// dim0/tile0/stride0 are in elements (data_size units).
// ---------------------------------------------------------------------------
__device__ __forceinline__ void tdm_load_tile_b16(
    const void* gptr, unsigned lds_addr, unsigned dim0, unsigned dim1,
    unsigned stride0, unsigned tile0, unsigned tile1) {
  const unsigned long long ga = (unsigned long long)(size_t)gptr;
  u32x4 g0;
  g0[0] = 1u;                                   // count=1 (valid), user mode
  g0[1] = lds_addr;                             // LDS byte address
  g0[2] = (unsigned)(ga & 0xffffffffu);         // global_addr[31:0]
  g0[3] = (unsigned)((ga >> 32) & 0x01ffffffu)  // global_addr[56:32]
          | (2u << 30);                         // type=2 ("image")
  i32x8 g1;
  g1[0] = (int)(1u << 16);                      // wg_mask=0, data_size=1 (2B)
  g1[1] = (int)((dim0 & 0xffffu) << 16);        // tensor_dim0[15:0]
  g1[2] = (int)((dim0 >> 16) | ((dim1 & 0xffffu) << 16));   // dim0[31:16]|dim1[15:0]
  g1[3] = (int)((dim1 >> 16) | ((tile0 & 0xffffu) << 16));  // dim1[31:16]|tile_dim0
  g1[4] = (int)(tile1 & 0xffffu);               // tile_dim1; tile_dim2=0
  g1[5] = (int)stride0;                         // dim0_stride[31:0]
  g1[6] = 0;                                    // dim0_stride[47:32]=0, dim1_stride[15:0]=0
  g1[7] = 0;                                    // dim1_stride[47:16]=0
  const i32x4 z4 = {0, 0, 0, 0};
#if __clang_major__ >= 23
  const i32x8 z8 = {0, 0, 0, 0, 0, 0, 0, 0};
  __builtin_amdgcn_tensor_load_to_lds(g0, g1, z4, z4, z8, 0);
#else
  __builtin_amdgcn_tensor_load_to_lds(g0, g1, z4, z4, 0);
#endif
}
#endif  // USE_TDM

__device__ __forceinline__ void wait_tensorcnt0() {
#if USE_TDM
  __builtin_amdgcn_s_wait_tensorcnt(0);
#endif
}

// ---------------------------------------------------------------------------
// f32 -> f16 conversion (weights / activations)
// ---------------------------------------------------------------------------
__global__ void cvt_f16_kernel(const float* __restrict__ src,
                               _Float16* __restrict__ dst, int n) {
  int i = blockIdx.x * blockDim.x + threadIdx.x;
  const int stride = gridDim.x * blockDim.x;
  for (; i < n; i += stride) dst[i] = (_Float16)src[i];
}

// ---------------------------------------------------------------------------
// Row LayerNorm (cols = 1024), f32 in -> f16 out. One block (256 thr) per row.
// ---------------------------------------------------------------------------
__global__ __launch_bounds__(256)
void layernorm_f16_kernel(const float* __restrict__ x, const float* __restrict__ w,
                          const float* __restrict__ b, _Float16* __restrict__ out,
                          int cols) {
  const int row = blockIdx.x;
  const float* xr = x + (size_t)row * cols;
  __shared__ float red[256];

  float s = 0.f;
  for (int i = threadIdx.x; i < cols; i += 256) s += xr[i];
  red[threadIdx.x] = s;
  __syncthreads();
  for (int o = 128; o > 0; o >>= 1) {
    if (threadIdx.x < o) red[threadIdx.x] += red[threadIdx.x + o];
    __syncthreads();
  }
  const float mu = red[0] / (float)cols;
  __syncthreads();

  s = 0.f;
  for (int i = threadIdx.x; i < cols; i += 256) {
    const float d = xr[i] - mu;
    s += d * d;
  }
  red[threadIdx.x] = s;
  __syncthreads();
  for (int o = 128; o > 0; o >>= 1) {
    if (threadIdx.x < o) red[threadIdx.x] += red[threadIdx.x + o];
    __syncthreads();
  }
  const float rstd = rsqrtf(red[0] / (float)cols + 1e-5f);

  _Float16* orow = out + (size_t)row * cols;
  for (int i = threadIdx.x; i < cols; i += 256)
    orow[i] = (_Float16)((xr[i] - mu) * rstd * w[i] + b[i]);
}

// ---------------------------------------------------------------------------
// Generic WMMA GEMM: C[M,N] = A[M,K](f16) * B[K,N](f16) + bias (+ GELU) (+ res)
// Block: 256 threads (8 waves). Block tile 128x128, wave tile 32x64 (8 WMMA
// per K-step), K-step 32, ping-pong double-buffered LDS tiles.
// A tile: TDM async load (one descriptor per block per K-step, wave 0 issues,
// TENSORcnt fenced before the pipeline barrier). B tile: register transpose.
// ---------------------------------------------------------------------------
template <bool GELU, bool RES, bool OUT32, bool OUT16>
__global__ __launch_bounds__(256)
void gemm_wmma_kernel(const _Float16* __restrict__ A, const _Float16* __restrict__ B,
                      const float* __restrict__ bias, const float* __restrict__ res,
                      float* __restrict__ out32, _Float16* __restrict__ out16,
                      int M, int N, int K) {
  __shared__ __align__(16) _Float16 As[2][128 * 32];  // A tiles, row-major
  __shared__ __align__(16) _Float16 Bt[2][128 * 32];  // B tiles, transposed [n][k]

  const int tid  = threadIdx.x;
  const int lane = tid & 31;
  const int warp = tid >> 5;
  const int mBase = blockIdx.y * 128;
  const int nBase = blockIdx.x * 128;
  const int m0 = (warp & 3) * 32;   // wave row offset within block tile
  const int n0 = (warp >> 2) * 64;  // wave col offset within block tile

  v8f acc[2][4] = {};

  // Stage one 128x32 A tile + one 32x128 B tile (transposed) into buffer 'buf'.
  auto stage = [&](int kk, int buf) {
#if USE_TDM
    // A tile via Tensor Data Mover: tile 32(x) x 128(y) halves of the A matrix,
    // row-major into As[buf] (= [128 rows][32 halves]). One issue per block.
    if (warp == 0) {
      tdm_load_tile_b16(A + (size_t)mBase * K + kk,
                        (unsigned)(size_t)(&As[buf][0]),
                        (unsigned)K, (unsigned)M, (unsigned)K,
                        32u, 128u);
    }
#else
    // A: 128 rows x 32 halves; 512 uint4 segments -> 2 per thread
#pragma unroll
    for (int it = 0; it < 2; ++it) {
      const int lin = tid + it * 256;
      const int r = lin >> 2, seg = lin & 3;
      const uint4 v = *(const uint4*)(A + (size_t)(mBase + r) * K + kk + seg * 8);
      *(uint4*)(&As[buf][r * 32 + seg * 8]) = v;
    }
#endif
    // B: 32 rows x 128 halves; 512 uint4 segments -> 2 per thread, transposed
#pragma unroll
    for (int it = 0; it < 2; ++it) {
      const int lin = tid + it * 256;
      const int r = lin >> 4, seg = lin & 15;  // r: k-row 0..31, seg: col group
      const uint4 v = *(const uint4*)(B + (size_t)(kk + r) * N + nBase + seg * 8);
      const _Float16* h = (const _Float16*)&v;
#pragma unroll
      for (int j = 0; j < 8; ++j) Bt[buf][(seg * 8 + j) * 32 + r] = h[j];
    }
  };

  stage(0, 0);
  wait_tensorcnt0();
  __syncthreads();

  const int nsteps = K / 32;
  for (int step = 0; step < nsteps; ++step) {
    const int buf = step & 1;
    if (step + 1 < nsteps) stage((step + 1) * 32, buf ^ 1);
    if (step + 2 < nsteps) {
      __builtin_prefetch(B + (size_t)((step + 2) * 32 + (tid >> 3)) * N + nBase, 0, 1);
    }

    const v16h a0 = frag_ld_lds(&As[buf][0], m0, lane);
    const v16h a1 = frag_ld_lds(&As[buf][0], m0 + 16, lane);
#pragma unroll
    for (int ni = 0; ni < 4; ++ni) {
      const v16h b = frag_ld_lds(&Bt[buf][0], n0 + ni * 16, lane);
      acc[0][ni] = wmma_f16(a0, b, acc[0][ni]);
      acc[1][ni] = wmma_f16(a1, b, acc[1][ni]);
    }
    wait_tensorcnt0();   // TDM for next buffer complete (wave 0; others pass)
    __syncthreads();     // next buffer staged; current reads complete
  }

  // ---- epilogue ----
  const int hi = lane >> 4, nn = lane & 15;
#pragma unroll
  for (int mi = 0; mi < 2; ++mi) {
#pragma unroll
    for (int ni = 0; ni < 4; ++ni) {
      const int col = nBase + n0 + ni * 16 + nn;
#pragma unroll
      for (int r = 0; r < 8; ++r) {
        const int row = mBase + m0 + mi * 16 + r + 8 * hi;
        float v = acc[mi][ni][r] + bias[col];
        if (GELU) v = gelu_tanh(v);
        if (RES) v += res[(size_t)row * N + col];
        if (OUT32) out32[(size_t)row * N + col] = v;
        if (OUT16) out16[(size_t)row * N + col] = (_Float16)v;
      }
    }
  }
}

// ---------------------------------------------------------------------------
// Stick-breaking attention. One wave per (16-query tile, head).
// qkv layout: [t][h*192 + {0:q, 64:k, 128:v}] (f16). q pre-scaled by 1/8.
// Walk 32-key tiles from the diagonal down to the sequence start; serial
// backward scan per query row carries the log(1-beta) suffix accumulator.
// ---------------------------------------------------------------------------
__global__ __launch_bounds__(32)
void sb_attn_kernel(const _Float16* __restrict__ qkv, const int* __restrict__ cu,
                    _Float16* __restrict__ oattn) {
  const int qbase = blockIdx.x * 16;
  const int head  = blockIdx.y;
  const int lane  = threadIdx.x;
  const int m = lane & 15, hi = lane >> 4;

  int seq_start = 0;
#pragma unroll
  for (int s = 0; s < 4; ++s)
    if (qbase >= cu[s]) seq_start = cu[s];

  __shared__ float    zbuf[16][32];
  __shared__ _Float16 abuf[16][32];
  __shared__ __align__(16) _Float16 vbuf[64][32];  // [d][s-local]

  // q fragments (rows qbase..qbase+15, d 0..31 and 32..63), scaled by 1/8
  v16h qa0, qa1;
  {
    const _Float16* qrow = qkv + (size_t)(qbase + m) * QKVW + head * 192;
#pragma unroll
    for (int j = 0; j < 8; ++j) {
      const int k = kpair(j, hi);
      qa0[2 * j]     = qrow[k]      * (_Float16)0.125f;
      qa0[2 * j + 1] = qrow[k + 1]  * (_Float16)0.125f;
      qa1[2 * j]     = qrow[k + 32] * (_Float16)0.125f;
      qa1[2 * j + 1] = qrow[k + 33] * (_Float16)0.125f;
    }
  }

  v8f oacc[4] = {};
  float accv = 0.f;  // per-row log prod(1-beta) carry (valid for lane < 16)

  const int ks_top = (qbase + 15) & ~31;
  for (int ks = ks_top; ks >= seq_start; ks -= 32) {
    // ---- z = q . k^T for this 16x32 tile (2 n-halves x 2 chained K-steps) --
#pragma unroll
    for (int nh = 0; nh < 2; ++nh) {
      const _Float16* krow =
          qkv + (size_t)(ks + nh * 16 + m) * QKVW + head * 192 + 64;
      v16h kb0, kb1;
#pragma unroll
      for (int j = 0; j < 8; ++j) {
        const int d = kpair(j, hi);
        kb0[2 * j]     = krow[d];
        kb0[2 * j + 1] = krow[d + 1];
        kb1[2 * j]     = krow[d + 32];
        kb1[2 * j + 1] = krow[d + 33];
      }
      v8f z = {};
      z = wmma_f16(qa0, kb0, z);
      z = wmma_f16(qa1, kb1, z);
#pragma unroll
      for (int r = 0; r < 8; ++r) zbuf[r + 8 * hi][nh * 16 + m] = z[r];
    }
    // ---- stage V tile transposed: vbuf[d][s-local] ----
    {
      const _Float16* vrow = qkv + (size_t)(ks + lane) * QKVW + head * 192 + 128;
#pragma unroll
      for (int j = 0; j < 64; ++j) vbuf[j][lane] = vrow[j];
    }
    __syncthreads();

    // ---- serial stick-breaking scan (16 lanes, one query row each) ----
    if (lane < 16) {
      const int i = qbase + lane;
      float a = accv;
      for (int c = 31; c >= 0; --c) {
        const int j = ks + c;
        float Aij = 0.f;
        if (j < i) {
          const float p   = zbuf[lane][c];
          const float sp  = (p > 20.f)  ? p  : log1pf(expf(p));    // softplus(p)
          const float spn = (p < -20.f) ? -p : log1pf(expf(-p));   // softplus(-p)
          Aij = expf(a - spn);  // sigmoid(p) * exp(suffix)
          a  -= sp;             // accumulate log(1-beta)
        }
        abuf[lane][c] = (_Float16)Aij;
      }
      accv = a;
    }
    __syncthreads();

    // ---- o += A(16x32) @ V(32x64): 4 WMMAs ----
    const v16h af = frag_ld_lds(&abuf[0][0], 0, lane);
#pragma unroll
    for (int dt = 0; dt < 4; ++dt) {
      const v16h vb = frag_ld_lds(&vbuf[0][0], dt * 16, lane);
      oacc[dt] = wmma_f16(af, vb, oacc[dt]);
    }
    __syncthreads();  // zbuf/abuf/vbuf reused next tile
  }

  // ---- write o (f16) into [T, HID] with head offset ----
#pragma unroll
  for (int dt = 0; dt < 4; ++dt) {
#pragma unroll
    for (int r = 0; r < 8; ++r) {
      const int row = qbase + r + 8 * hi;
      const int col = head * HD + dt * 16 + m;
      oattn[(size_t)row * HID + col] = (_Float16)oacc[dt][r];
    }
  }
}

// ---------------------------------------------------------------------------
// Host-side orchestration
// ---------------------------------------------------------------------------
extern "C" void kernel_launch(void* const* d_in, const int* in_sizes, int n_in,
                              void* d_out, int out_size, void* d_ws, size_t ws_size,
                              hipStream_t stream) {
  (void)in_sizes; (void)n_in; (void)out_size; (void)ws_size;

  const float* hidden   = (const float*)d_in[0];
  const int*   cu       = (const int*)d_in[1];
  const float* ln1_w    = (const float*)d_in[2];
  const float* ln1_b    = (const float*)d_in[3];
  const float* c_attn_w = (const float*)d_in[4];
  const float* c_attn_b = (const float*)d_in[5];
  const float* c_proj_w = (const float*)d_in[6];
  const float* c_proj_b = (const float*)d_in[7];
  const float* ln2_w    = (const float*)d_in[8];
  const float* ln2_b    = (const float*)d_in[9];
  const float* fc_w     = (const float*)d_in[10];
  const float* fc_b     = (const float*)d_in[11];
  const float* proj_w   = (const float*)d_in[12];
  const float* proj_b   = (const float*)d_in[13];
  float* out = (float*)d_out;

  // workspace carve-out (256B aligned)
  char* ws = (char*)d_ws;
  size_t off = 0;
  auto take = [&](size_t bytes) -> void* {
    void* p = ws + off;
    off = (off + bytes + 255) & ~(size_t)255;
    return p;
  };
  _Float16* wAttn16 = (_Float16*)take((size_t)HID * QKVW * 2);
  _Float16* wCprj16 = (_Float16*)take((size_t)HID * HID * 2);
  _Float16* wFc16   = (_Float16*)take((size_t)HID * INNER * 2);
  _Float16* wPrj16  = (_Float16*)take((size_t)INNER * HID * 2);
  _Float16* xln1    = (_Float16*)take((size_t)T_TOK * HID * 2);
  _Float16* qkv16   = (_Float16*)take((size_t)T_TOK * QKVW * 2);
  _Float16* oattn   = (_Float16*)take((size_t)T_TOK * HID * 2);
  float*    hbuf    = (float*)   take((size_t)T_TOK * HID * 4);
  _Float16* xln2    = (_Float16*)take((size_t)T_TOK * HID * 2);
  _Float16* gact    = (_Float16*)take((size_t)T_TOK * INNER * 2);

  // 1) weight conversions to f16
  cvt_f16_kernel<<<4096, 256, 0, stream>>>(c_attn_w, wAttn16, HID * QKVW);
  cvt_f16_kernel<<<2048, 256, 0, stream>>>(c_proj_w, wCprj16, HID * HID);
  cvt_f16_kernel<<<4096, 256, 0, stream>>>(fc_w,     wFc16,   HID * INNER);
  cvt_f16_kernel<<<4096, 256, 0, stream>>>(proj_w,   wPrj16,  INNER * HID);

  // 2) LN1
  layernorm_f16_kernel<<<T_TOK, 256, 0, stream>>>(hidden, ln1_w, ln1_b, xln1, HID);

  // 3) QKV GEMM: [2048,1024] x [1024,3072] -> f16 qkv
  gemm_wmma_kernel<false, false, false, true>
      <<<dim3(QKVW / 128, T_TOK / 128), 256, 0, stream>>>(
          xln1, wAttn16, c_attn_b, nullptr, nullptr, qkv16, T_TOK, QKVW, HID);

  // 4) stick-breaking attention
  sb_attn_kernel<<<dim3(T_TOK / 16, NHEAD), 32, 0, stream>>>(qkv16, cu, oattn);

  // 5) c_proj GEMM + residual(hidden) -> h (f32)
  gemm_wmma_kernel<false, true, true, false>
      <<<dim3(HID / 128, T_TOK / 128), 256, 0, stream>>>(
          oattn, wCprj16, c_proj_b, hidden, hbuf, nullptr, T_TOK, HID, HID);

  // 6) LN2
  layernorm_f16_kernel<<<T_TOK, 256, 0, stream>>>(hbuf, ln2_w, ln2_b, xln2, HID);

  // 7) FC GEMM + GELU -> f16
  gemm_wmma_kernel<true, false, false, true>
      <<<dim3(INNER / 128, T_TOK / 128), 256, 0, stream>>>(
          xln2, wFc16, fc_b, nullptr, nullptr, gact, T_TOK, INNER, HID);

  // 8) down-proj GEMM + residual(h) -> out (f32)
  gemm_wmma_kernel<false, true, true, false>
      <<<dim3(HID / 128, T_TOK / 128), 256, 0, stream>>>(
          gact, wPrj16, proj_b, hbuf, out, nullptr, T_TOK, HID, INNER);
}
